// Memory_MultiheadAttention_40132174414424
// MI455X (gfx1250) — matrix operand order repaired and checked by reading, not compiled, not run on hardware
//
#include <hip/hip_runtime.h>
#include <hip/hip_bf16.h>

// Problem constants: B=4,L=64,E=8,Q=16,K=32,H=512,NH=8,d=64
#define NGROUP 2048          // B*L*E
#define HH     512
#define OUT0   16777216      // NGROUP*16*512 (first output segment)
#define SCALE  0.125f        // 1/sqrt(64)
#define WSTR   40            // padded row stride (bf16): 16B aligned, bank-conflict free
#define XSTR   520           // sQ/sK/sX row stride (bf16)
#define WTILE  16384         // elems per (weight, kc) staged tile: 512*32
#define WMAT   262144        // elems per transposed bf16 weight: 512*512

typedef __bf16 bf16;
typedef __attribute__((ext_vector_type(16))) bf16  bf16x16;
typedef __attribute__((ext_vector_type(8)))  bf16  bf16x8;
typedef __attribute__((ext_vector_type(8)))  float v8f;
typedef int v4i __attribute__((__vector_size__(16)));
typedef __attribute__((address_space(1))) v4i* as1_v4i_ptr;
typedef __attribute__((address_space(3))) v4i* as3_v4i_ptr;

#define WMMA_BF16(a, b, c) \
  __builtin_amdgcn_wmma_f32_16x16x32_bf16(false, (a), false, (b), (short)0, (c), false, false)

#if defined(__has_builtin)
#if __has_builtin(__builtin_amdgcn_global_load_async_to_lds_b128)
#define HAVE_ASYNC 1
#endif
#endif
#ifndef HAVE_ASYNC
#define HAVE_ASYNC 0
#endif

__device__ __forceinline__ v8f vzero() {
  v8f z;
#pragma unroll
  for (int i = 0; i < 8; ++i) z[i] = 0.0f;
  return z;
}

__device__ __forceinline__ bf16x16 ld16(const bf16* p0, const bf16* p1) {
  union { bf16x16 v; bf16x8 h[2]; } u;
  u.h[0] = *(const bf16x8*)p0;
  u.h[1] = *(const bf16x8*)(p1);
  return u.v;
}
// A-fragment (16x32): lane<16 holds K {kb..kb+15}; lane>=16 holds K {kb+16..kb+31}
__device__ __forceinline__ bf16x16 afrag_lds(const bf16* row, int kbase, int hi) {
  const bf16* p = row + kbase + hi * 16;
  return ld16(p, p + 8);
}
__device__ __forceinline__ bf16x16 afrag_glb(const float* row, int kbase, int hi) {
  bf16x16 r;
  const float* p0 = row + kbase + hi * 16;
  const float* p1 = p0 + 8;
#pragma unroll
  for (int i = 0; i < 8; ++i) { r[i] = (bf16)p0[i]; r[8 + i] = (bf16)p1[i]; }
  return r;
}
__device__ __forceinline__ bf16x16 bfrag_lds(const bf16* p) { return ld16(p, p + 8); }

// Stage one 512x32 bf16 weight tile (tile-major in global) into LDS with padding.
// 2048 16B chunks / 256 threads = 8 per thread. Async path uses ASYNCcnt.
__device__ __forceinline__ void issue_tile(const bf16* __restrict__ g, bf16* l, int tid) {
#pragma unroll
  for (int it = 0; it < 8; ++it) {
    int c = tid + (it << 8);          // chunk 0..2047
    int n = c >> 2, q = c & 3;        // row n, 16B quarter q
#if HAVE_ASYNC
    __builtin_amdgcn_global_load_async_to_lds_b128(
        (as1_v4i_ptr)(g + n * 32 + q * 8),
        (as3_v4i_ptr)(l + n * WSTR + q * 8), 0, 0);
#else
    *(bf16x8*)(l + n * WSTR + q * 8) = *(const bf16x8*)(g + n * 32 + q * 8);
#endif
  }
}
__device__ __forceinline__ void wait_async8() {
#if HAVE_ASYNC
  asm volatile("s_wait_asynccnt 0x8" ::: "memory");
#endif
}
__device__ __forceinline__ void wait_async0() {
#if HAVE_ASYNC
  asm volatile("s_wait_asynccnt 0x0" ::: "memory");
#endif
}

// C = A[16|32 x 512] * W[512x512] over K=512; W^T pre-converted bf16 tile-major in gWt.
// Double-buffered async staging: tile kc+1 loads while tile kc feeds WMMA.
template <int NTW, bool ALDS>
__device__ __forceinline__ void gemm_k512(const float* gA, const bf16* sA, int mrow,
                                          const bf16* __restrict__ gWt,
                                          bf16* sW0, bf16* sW1,
                                          int ntile0, int tid, int lane, v8f* acc) {
  const int l16 = lane & 15, hi = lane >> 4;
  __syncthreads();                       // phase entry: prior users of sW0/sW1 done
  issue_tile(gWt, sW0, tid);
  for (int kc = 0; kc < 16; ++kc) {
    bf16* cur = (kc & 1) ? sW1 : sW0;
    bf16* nxt = (kc & 1) ? sW0 : sW1;
    if (kc < 15) { issue_tile(gWt + (kc + 1) * WTILE, nxt, tid); wait_async8(); }
    else         { wait_async0(); }
    __syncthreads();                     // everyone's tile-kc chunks landed
    bf16x16 a;
    if (ALDS) a = afrag_lds(sA + mrow * XSTR, kc * 32, hi);
    else      a = afrag_glb(gA + mrow * HH,   kc * 32, hi);
#pragma unroll
    for (int j = 0; j < NTW; ++j) {
      bf16x16 b = bfrag_lds(cur + ((ntile0 + j) * 16 + l16) * WSTR + hi * 16);
      acc[j] = WMMA_BF16(a, b, acc[j]);
    }
    __syncthreads();                     // reads of 'cur' done before it is overwritten
  }
}

// ---- Prologue: W (fp32 [k][n]) -> ws (bf16, tile-major W^T: [kc][n][32]) ----
__global__ __launch_bounds__(256) void transpose_weights_kernel(
    const float* __restrict__ Wq, const float* __restrict__ Wk,
    const float* __restrict__ Wv, const float* __restrict__ Wo,
    bf16* __restrict__ wsW) {
  __shared__ __align__(16) bf16 sT[HH * WSTR];
  const int m  = blockIdx.x >> 4;        // weight index 0..3
  const int kc = blockIdx.x & 15;        // k-chunk 0..15
  const int tid = threadIdx.x;
  const float* W = (m == 0) ? Wq : (m == 1) ? Wk : (m == 2) ? Wv : Wo;
  const float* src = W + kc * (32 * HH);
#pragma unroll
  for (int j = 0; j < 16; ++j) {         // coalesced read, transposed LDS write
    int i4 = (tid + (j << 8)) << 2;
    float4 wv = *(const float4*)(src + i4);
    int n = i4 & 511, k = i4 >> 9;
    bf16* d = sT + n * WSTR + k;
    d[0]        = (bf16)wv.x;
    d[WSTR]     = (bf16)wv.y;
    d[2 * WSTR] = (bf16)wv.z;
    d[3 * WSTR] = (bf16)wv.w;
  }
  __syncthreads();
  bf16* dst = wsW + m * WMAT + kc * WTILE;
#pragma unroll
  for (int j = 0; j < 8; ++j) {          // contiguous 16B writes
    int c = tid + (j << 8);
    int n = c >> 2, q = c & 3;
    *(bf16x8*)(dst + n * 32 + q * 8) = *(const bf16x8*)(sT + n * WSTR + q * 8);
  }
}

__global__ __launch_bounds__(256) void mha_fused_kernel(
    const float* __restrict__ query, const float* __restrict__ key,
    const float* __restrict__ value, const float* __restrict__ bias,
    const bf16* __restrict__ wsW,
    const float* __restrict__ bq, const float* __restrict__ bk,
    const float* __restrict__ bv, const float* __restrict__ bo,
    float* __restrict__ out) {
  __shared__ __align__(16) bf16 sW0[HH * WSTR];     // weight tile ping (40960 B)
  __shared__ __align__(16) bf16 sW1[HH * WSTR];     // weight tile pong (40960 B)
  __shared__ __align__(16) bf16 sQ[16 * XSTR];      // q proj (scaled), row-major
  __shared__ __align__(16) bf16 sK[32 * XSTR];      // k proj, row-major
  __shared__ __align__(16) bf16 sVt[HH * WSTR];     // v proj, transposed [h][kk]
  __shared__ __align__(16) bf16 sX[16 * XSTR];      // attn output, combined heads
  __shared__ __align__(16) bf16 sPb[8 * 16 * WSTR]; // attn probs (bf16) per head
  float* sPtop = (float*)sW0;                       // fp32 probs heads {0,4}; sW0 idle here

  const int g    = blockIdx.x;     // (b*L + l)*E + e
  const int tid  = threadIdx.x;
  const int w    = tid >> 5;
  const int lane = tid & 31;
  const int l16  = lane & 15, hi = lane >> 4;

  const float* gQ = query + (size_t)g * (16 * HH);
  const float* gK = key   + (size_t)g * (32 * HH);
  const float* gV = value + (size_t)g * (32 * HH);

  // ---- Q projection: sQ = (query*Wq + bq) * scale ----
  {
    v8f acc[4];
#pragma unroll
    for (int j = 0; j < 4; ++j) acc[j] = vzero();
    gemm_k512<4, false>(gQ, sX, l16, wsW + 0 * WMAT, sW0, sW1, w * 4, tid, lane, acc);
#pragma unroll
    for (int j = 0; j < 4; ++j) {
      int col = (w * 4 + j) * 16 + l16;
      float bb = bq[col];
#pragma unroll
      for (int r = 0; r < 8; ++r)
        sQ[(r + 8 * hi) * XSTR + col] = (bf16)((acc[j][r] + bb) * SCALE);
    }
  }
  // ---- K projection: sK = key*Wk + bk ----
  {
    v8f acc[8];
#pragma unroll
    for (int j = 0; j < 8; ++j) acc[j] = vzero();
    const int mt = w & 1, nt0 = (w >> 1) * 8;
    gemm_k512<8, false>(gK, sX, mt * 16 + l16, wsW + 1 * WMAT, sW0, sW1, nt0, tid, lane, acc);
#pragma unroll
    for (int j = 0; j < 8; ++j) {
      int col = (nt0 + j) * 16 + l16;
      float bb = bk[col];
#pragma unroll
      for (int r = 0; r < 8; ++r)
        sK[(mt * 16 + r + 8 * hi) * XSTR + col] = (bf16)(acc[j][r] + bb);
    }
  }
  // ---- V projection (stored transposed): sVt[h][kk] ----
  {
    v8f acc[8];
#pragma unroll
    for (int j = 0; j < 8; ++j) acc[j] = vzero();
    const int mt = w & 1, nt0 = (w >> 1) * 8;
    gemm_k512<8, false>(gV, sX, mt * 16 + l16, wsW + 2 * WMAT, sW0, sW1, nt0, tid, lane, acc);
#pragma unroll
    for (int j = 0; j < 8; ++j) {
      int col = (nt0 + j) * 16 + l16;
      float bb = bv[col];
#pragma unroll
      for (int r = 0; r < 8; ++r)
        sVt[col * WSTR + (mt * 16 + r + 8 * hi)] = (bf16)(acc[j][r] + bb);
    }
  }
  __syncthreads();

  // ---- Attention: wave w = head n; logits = q_n k_n^T + bias; softmax over K=32 ----
  {
    const int n = w;
    bf16x16 qa0 = afrag_lds(sQ + l16 * XSTR, n * 64, hi);
    bf16x16 qa1 = afrag_lds(sQ + l16 * XSTR, n * 64 + 32, hi);
    const float* bptr = bias + (size_t)g * 512;
    v8f c0, c1;
#pragma unroll
    for (int r = 0; r < 8; ++r) {
      c0[r] = bptr[(r + 8 * hi) * 32 + l16];
      c1[r] = bptr[(r + 8 * hi) * 32 + 16 + l16];
    }
    bf16x16 kb;
    kb = bfrag_lds(sK + l16 * XSTR + n * 64 + hi * 16);             c0 = WMMA_BF16(qa0, kb, c0);
    kb = bfrag_lds(sK + l16 * XSTR + n * 64 + 32 + hi * 16);        c0 = WMMA_BF16(qa1, kb, c0);
    kb = bfrag_lds(sK + (16 + l16) * XSTR + n * 64 + hi * 16);      c1 = WMMA_BF16(qa0, kb, c1);
    kb = bfrag_lds(sK + (16 + l16) * XSTR + n * 64 + 32 + hi * 16); c1 = WMMA_BF16(qa1, kb, c1);

#pragma unroll
    for (int r = 0; r < 8; ++r) {        // row softmax: 16-lane reductions
      float m = fmaxf(c0[r], c1[r]);
      m = fmaxf(m, __shfl_xor(m, 1, 32));
      m = fmaxf(m, __shfl_xor(m, 2, 32));
      m = fmaxf(m, __shfl_xor(m, 4, 32));
      m = fmaxf(m, __shfl_xor(m, 8, 32));
      float p0 = __expf(c0[r] - m), p1 = __expf(c1[r] - m);
      float s = p0 + p1;
      s += __shfl_xor(s, 1, 32);
      s += __shfl_xor(s, 2, 32);
      s += __shfl_xor(s, 4, 32);
      s += __shfl_xor(s, 8, 32);
      float inv = 1.0f / s;
      p0 *= inv; p1 *= inv;
      int row = r + 8 * hi;
      sPb[(n * 16 + row) * WSTR + l16]      = (bf16)p0;
      sPb[(n * 16 + row) * WSTR + 16 + l16] = (bf16)p1;
      if ((n & 3) == 0) {                // only heads 0 and 4 feed top_attn
        sPtop[(n >> 2) * 512 + row * 32 + l16]      = p0;
        sPtop[(n >> 2) * 512 + row * 32 + 16 + l16] = p1;
      }
    }
  }
  __syncthreads();

  // ---- top_attn raw-view slice -> out[OUT0 + ...] ----
  {
    const size_t topBase = (size_t)OUT0 + (size_t)(g >> 3) * 4096;
    const int eg = g & 7;
#pragma unroll
    for (int jj = 0; jj < 2; ++jj) {
      int i = tid + (jj << 8);
      int t = i >> 8, rem = i & 255;
      int kp = rem >> 3, ep = rem & 7;
      float v = sPtop[t * 512 + (kp >> 2) * 32 + ((kp & 3) << 3) + ep];
      out[topBase + (size_t)(2 * eg + t) * 256 + rem] = v;
    }
  }

  // ---- x_n = attn_n * v_n -> sX (combined heads) ----
  {
    const int n = w;
    bf16x16 pa = afrag_lds(sPb + (n * 16 + l16) * WSTR, 0, hi);
#pragma unroll
    for (int dt = 0; dt < 4; ++dt) {
      bf16x16 vb = bfrag_lds(sVt + (n * 64 + dt * 16 + l16) * WSTR + hi * 16);
      v8f x = vzero();
      x = WMMA_BF16(pa, vb, x);
#pragma unroll
      for (int r = 0; r < 8; ++r)
        sX[(r + 8 * hi) * XSTR + n * 64 + dt * 16 + l16] = (bf16)x[r];
    }
  }

  // ---- Output projection: out = sX*Wo + bo ----
  {
    v8f acc[4];
#pragma unroll
    for (int j = 0; j < 4; ++j) acc[j] = vzero();
    gemm_k512<4, true>(gQ, sX, l16, wsW + 3 * WMAT, sW0, sW1, w * 4, tid, lane, acc);
    float* gO = out + (size_t)g * (16 * HH);
#pragma unroll
    for (int j = 0; j < 4; ++j) {
      int col = (w * 4 + j) * 16 + l16;
      float bb = bo[col];
#pragma unroll
      for (int r = 0; r < 8; ++r)
        gO[(r + 8 * hi) * HH + col] = acc[j][r] + bb;
    }
  }
}

extern "C" void kernel_launch(void* const* d_in, const int* in_sizes, int n_in,
                              void* d_out, int out_size, void* d_ws, size_t ws_size,
                              hipStream_t stream) {
  const float* query = (const float*)d_in[0];
  const float* key   = (const float*)d_in[1];
  const float* value = (const float*)d_in[2];
  const float* bias  = (const float*)d_in[3];
  const float* Wq    = (const float*)d_in[4];
  const float* bq    = (const float*)d_in[5];
  const float* Wk    = (const float*)d_in[6];
  const float* bk    = (const float*)d_in[7];
  const float* Wv    = (const float*)d_in[8];
  const float* bv    = (const float*)d_in[9];
  const float* Wo    = (const float*)d_in[10];
  const float* bo    = (const float*)d_in[11];
  float* out = (float*)d_out;
  bf16* wsW  = (bf16*)d_ws;   // 4*512*512 bf16 = 2 MB

  hipLaunchKernelGGL(transpose_weights_kernel, dim3(64), dim3(256), 0, stream,
                     Wq, Wk, Wv, Wo, wsW);
  hipLaunchKernelGGL(mha_fused_kernel, dim3(NGROUP), dim3(256), 0, stream,
                     query, key, value, bias, wsW, bq, bk, bv, bo, out);
}
